// KGPolicy_13408887898392
// MI455X (gfx1250) — matrix operand order repaired and checked by reading, not compiled, not run on hardware
//
#include <hip/hip_runtime.h>
#include <hip/hip_bf16.h>

// ---------------- problem constants (match reference) ----------------
#define N_ENT     100000
#define N_ITEMS   50000
#define EDGE_T    32
#define BATCH     8192
#define NUM_SAMPLE 16
#define D_IN      64
#define D_HID     32
#define SLOPE     0.01f
#define ITEM_HI   (N_ITEMS - 1)

typedef __attribute__((ext_vector_type(16))) __bf16          v16bf;
typedef __attribute__((ext_vector_type(8)))  float           v8f;
typedef __attribute__((ext_vector_type(8)))  unsigned short  v8us;
typedef __attribute__((ext_vector_type(16))) unsigned short  v16us;

// ---------------- small helpers ----------------
__device__ __forceinline__ unsigned short f2bfu(float f) {
    union { float f; unsigned u; } v; v.f = f;
    unsigned r = v.u + 0x7fffu + ((v.u >> 16) & 1u);   // round-to-nearest-even
    return (unsigned short)(r >> 16);
}
__device__ __forceinline__ float leaky(float x) { return x > 0.f ? x : SLOPE * x; }

// A-operand for v_wmma_f32_16x16x32_bf16 (ISA 7.12.2, 16-bit A 16x32):
// lane L holds row M=L&15; its 16 bf16 elements are two contiguous 8-element
// runs at K = K0+khalf and K0+khalf+16  ->  two 16-byte vector loads.
__device__ __forceinline__ v16bf loadA_vec(const unsigned short* rowptr, int K0, int khalf) {
    v8us lo = *(const v8us*)(rowptr + K0 + khalf);
    v8us hi = *(const v8us*)(rowptr + K0 + khalf + 16);
    v16us u = __builtin_shufflevector(lo, hi, 0, 1, 2, 3, 4, 5, 6, 7,
                                              8, 9, 10, 11, 12, 13, 14, 15);
    return __builtin_bit_cast(v16bf, u);
}
// B-operand: preswizzled pack, 16 contiguous ushorts per (tile,lane) = 1 vector load.
__device__ __forceinline__ v16bf loadB_pack(const unsigned short* pack, int tile, int lane) {
    return *(const v16bf*)(pack + (((size_t)tile * 32 + lane) << 4));
}
__device__ __forceinline__ v8f wmma_bf16(v16bf a, v16bf b, v8f c) {
    return __builtin_amdgcn_wmma_f32_16x16x32_bf16(false, a, false, b, (short)0, c, false, false);
}

__device__ __forceinline__ float wave_max(float v) {
#pragma unroll
    for (int o = 16; o; o >>= 1) v = fmaxf(v, __shfl_xor(v, o));
    return v;
}
__device__ __forceinline__ float wave_sum(float v) {
#pragma unroll
    for (int o = 16; o; o >>= 1) v += __shfl_xor(v, o);
    return v;
}
// argmax with first-index tie-break (matches jnp.argmax / stable argsort head)
__device__ __forceinline__ void wave_argmax(float& v, int& i) {
#pragma unroll
    for (int o = 16; o; o >>= 1) {
        float ov = __shfl_xor(v, o);
        int   oi = __shfl_xor(i, o);
        if (ov > v || (ov == v && oi < i)) { v = ov; i = oi; }
    }
}
__device__ __forceinline__ unsigned hashu(unsigned x) {
    x ^= x >> 17; x *= 0xed5ad4bbu; x ^= x >> 11;
    x *= 0xac4c1b51u; x ^= x >> 15; x *= 0x31848babu; x ^= x >> 14;
    return x;
}

// ======= one-time weight prepack: f32 -> bf16 in the ISA B-matrix layout ==========
// Layer1 tiles: (ks,nt) -> tile=ks*2+nt; k = ks*32 + (lane>>4)*8 + (j&7) + (j>>3)*16, n = (lane&15)+nt*16
// Layer2 tiles: nt in 0..3;              k = (lane>>4)*8 + (j&7) + (j>>3)*16,         n = (lane&15)+nt*16
__global__ void prepB(const float* __restrict__ W1r, const float* __restrict__ W1n,
                      const float* __restrict__ W2r, const float* __restrict__ W2n,
                      unsigned short* __restrict__ p1r, unsigned short* __restrict__ p1n,
                      unsigned short* __restrict__ p2r, unsigned short* __restrict__ p2n) {
    int t = blockIdx.x * blockDim.x + threadIdx.x;      // 2048 threads
    if (t >= 2048) return;
    int j    = t & 15;
    int lane = (t >> 4) & 31;
    int tile = t >> 9;                                  // 0..3
    {   // layer 1 (K=64: ks = tile>>1, nt = tile&1)
        int ks = tile >> 1, nt = tile & 1;
        int k = ks * 32 + (lane >> 4) * 8 + (j & 7) + (j >> 3) * 16;
        int n = (lane & 15) + nt * 16;
        p1r[t] = f2bfu(W1r[k * D_HID + n]);
        p1n[t] = f2bfu(W1n[k * D_HID + n]);
    }
    {   // layer 2 (K=32: nt = tile)
        int k = (lane >> 4) * 8 + (j & 7) + (j >> 3) * 16;
        int n = (lane & 15) + tile * 16;
        p2r[t] = f2bfu(W2r[k * D_IN + n]);
        p2n[t] = f2bfu(W2n[k * D_IN + n]);
    }
}

// ================= graph aggregation (segment mean, L2-resident atomics) ==========
__global__ void scatter64(const float* __restrict__ x, const int* __restrict__ edge,
                          float* __restrict__ aggr, float* __restrict__ cnt) {
    int tid = blockIdx.x * blockDim.x + threadIdx.x;   // N_ENT*EDGE_T*64 threads
    int e = tid >> 6, d = tid & 63;
    int src = e >> 5;
    int dst = edge[e];
    atomicAdd(&aggr[dst * 64 + d], x[src * 64 + d]);
    if (d == 0) atomicAdd(&cnt[dst], 1.0f);
}
__global__ void scatter32(const float* __restrict__ h, const int* __restrict__ edge,
                          float* __restrict__ aggr) {
    int tid = blockIdx.x * blockDim.x + threadIdx.x;   // N_ENT*EDGE_T*32 threads
    int e = tid >> 5, d = tid & 31;
    int src = e >> 5;
    int dst = edge[e];
    atomicAdd(&aggr[dst * 32 + d], h[src * 32 + d]);
}

// ================= f32 -> bf16 staging (mean division fused) =====================
__global__ void cvt64(const float* __restrict__ x, const float* __restrict__ aggr,
                      const float* __restrict__ cnt,
                      unsigned short* __restrict__ xbf, unsigned short* __restrict__ abf) {
    int i = blockIdx.x * blockDim.x + threadIdx.x;     // N_ENT*64
    float c = fmaxf(cnt[i >> 6], 1.0f);
    xbf[i] = f2bfu(x[i]);
    abf[i] = f2bfu(aggr[i] / c);
}
__global__ void cvt32(const float* __restrict__ h, const float* __restrict__ aggr,
                      const float* __restrict__ cnt,
                      unsigned short* __restrict__ hbf, unsigned short* __restrict__ abf) {
    int i = blockIdx.x * blockDim.x + threadIdx.x;     // N_ENT*32
    float c = fmaxf(cnt[i >> 5], 1.0f);
    hbf[i] = f2bfu(h[i]);
    abf[i] = f2bfu(aggr[i] / c);
}

// ================= layer 1: h1 = leaky(aggr@W1n + x@W1r + b1)  [100000,64]->[.,32]
__global__ void gemm1(const unsigned short* __restrict__ xbf,
                      const unsigned short* __restrict__ abf,
                      const unsigned short* __restrict__ p1r,
                      const unsigned short* __restrict__ p1n,
                      const float* __restrict__ b1, float* __restrict__ h1) {
    int wv   = (blockIdx.x * blockDim.x + threadIdx.x) >> 5;
    int lane = threadIdx.x & 31;
    if (wv >= N_ENT / 16) return;
    int row0  = wv << 4;
    int rA    = row0 + (lane & 15);
    int khalf = (lane >> 4) * 8;
    int ncol  = lane & 15;

    v8f acc[2];
#pragma unroll
    for (int nt = 0; nt < 2; ++nt) {
        float bb = b1[ncol + nt * 16];
#pragma unroll
        for (int r = 0; r < 8; ++r) acc[nt][r] = bb;
    }
#pragma unroll
    for (int ks = 0; ks < 2; ++ks) {
        int K0 = ks * 32;
        v16bf ax = loadA_vec(xbf + (size_t)rA * 64, K0, khalf);
        v16bf aa = loadA_vec(abf + (size_t)rA * 64, K0, khalf);
#pragma unroll
        for (int nt = 0; nt < 2; ++nt) {
            v16bf br = loadB_pack(p1r, ks * 2 + nt, lane);
            v16bf bn = loadB_pack(p1n, ks * 2 + nt, lane);
            acc[nt] = wmma_bf16(ax, br, acc[nt]);
            acc[nt] = wmma_bf16(aa, bn, acc[nt]);
        }
    }
    int mbase = (lane >> 4) * 8;
#pragma unroll
    for (int nt = 0; nt < 2; ++nt)
#pragma unroll
        for (int r = 0; r < 8; ++r)
            h1[(size_t)(row0 + mbase + r) * 32 + nt * 16 + ncol] = leaky(acc[nt][r]);
}

// ===== layer 2: gcn = rownorm(aggr2@W2n + h1@W2r + b2)  [100000,32]->[.,64] ======
__global__ void gemm2(const unsigned short* __restrict__ hbf,
                      const unsigned short* __restrict__ abf,
                      const unsigned short* __restrict__ p2r,
                      const unsigned short* __restrict__ p2n,
                      const float* __restrict__ b2, float* __restrict__ gcn) {
    int wv   = (blockIdx.x * blockDim.x + threadIdx.x) >> 5;
    int lane = threadIdx.x & 31;
    if (wv >= N_ENT / 16) return;
    int row0  = wv << 4;
    int rA    = row0 + (lane & 15);
    int khalf = (lane >> 4) * 8;
    int ncol  = lane & 15;

    v8f acc[4];
#pragma unroll
    for (int nt = 0; nt < 4; ++nt) {
        float bb = b2[ncol + nt * 16];
#pragma unroll
        for (int r = 0; r < 8; ++r) acc[nt][r] = bb;
    }
    v16bf ah = loadA_vec(hbf + (size_t)rA * 32, 0, khalf);
    v16bf aa = loadA_vec(abf + (size_t)rA * 32, 0, khalf);
#pragma unroll
    for (int nt = 0; nt < 4; ++nt) {
        v16bf br = loadB_pack(p2r, nt, lane);
        v16bf bn = loadB_pack(p2n, nt, lane);
        acc[nt] = wmma_bf16(ah, br, acc[nt]);
        acc[nt] = wmma_bf16(aa, bn, acc[nt]);
    }
    // fused L2 row-normalization: row m's 64 cols live in one 16-lane half across
    // the 4 n-tiles; xor masks 1..8 reduce within a half (C/D layout, ISA 7.12.2).
    int mbase = (lane >> 4) * 8;
#pragma unroll
    for (int r = 0; r < 8; ++r) {
        float ss = 0.f;
#pragma unroll
        for (int nt = 0; nt < 4; ++nt) ss += acc[nt][r] * acc[nt][r];
#pragma unroll
        for (int o = 1; o < 16; o <<= 1) ss += __shfl_xor(ss, o);
        float inv = 1.0f / fmaxf(sqrtf(ss), 1e-12f);
#pragma unroll
        for (int nt = 0; nt < 4; ++nt)
            gcn[(size_t)(row0 + mbase + r) * 64 + nt * 16 + ncol] = acc[nt][r] * inv;
    }
}

// ---------------- fused attention dot: sum_d leaky(pos*e) * u  (float4 loads) ----
__device__ __forceinline__ float att_dot(const float* __restrict__ gcn,
                                         int pe, int e, int u) {
    const float4* pv = (const float4*)(gcn + (size_t)pe * 64);
    const float4* ev = (const float4*)(gcn + (size_t)e * 64);
    const float4* uv = (const float4*)(gcn + (size_t)u * 64);
    float p = 0.f;
#pragma unroll
    for (int d = 0; d < 16; ++d) {
        float4 a = pv[d], b = ev[d], c = uv[d];
        p += leaky(a.x * b.x) * c.x + leaky(a.y * b.y) * c.y +
             leaky(a.z * b.z) * c.z + leaky(a.w * b.w) * c.w;
    }
    return p;
}

// ================= kg_step stage 1: softmax over 32 neighbors, argmax ============
__global__ void kg1(const float* __restrict__ gcn, const int* __restrict__ adj,
                    const int* __restrict__ pos, const int* __restrict__ u_id,
                    int* __restrict__ one_hop, float* __restrict__ one_log) {
    int b    = (blockIdx.x * blockDim.x + threadIdx.x) >> 5;  // BATCH waves exactly
    int lane = threadIdx.x & 31;
    int u  = u_id[b];
    int pe = pos[b];
    int e  = adj[pe * EDGE_T + lane];
    __builtin_prefetch(gcn + (size_t)e * 64, 0, 1);
    float p   = att_dot(gcn, pe, e, u);
    float mx  = wave_max(p);
    float sum = wave_sum(__expf(p - mx));
    float lsm = p - mx - __logf(sum);     // log softmax
    float bv = p; int bi = lane;
    wave_argmax(bv, bi);                   // winner lane index (all lanes agree)
    if (lane == 0) {
        one_hop[b] = adj[pe * EDGE_T + bi];
        one_log[b] = __shfl(lsm, bi);
    }
}

// ======= kg_step stage 2: top-16 select + filter + discriminator prune ===========
__global__ void kg2(const float* __restrict__ gcn, const int* __restrict__ adj,
                    const int* __restrict__ hop, const int* __restrict__ u_id,
                    const float* __restrict__ dis_u, const float* __restrict__ dis_i,
                    const float* __restrict__ one_log, int step,
                    int* __restrict__ pos_next, float* __restrict__ out) {
    int b    = (blockIdx.x * blockDim.x + threadIdx.x) >> 5;
    int lane = threadIdx.x & 31;
    int u  = u_id[b];
    int pe = hop[b];
    int e  = adj[pe * EDGE_T + lane];
    float p   = att_dot(gcn, pe, e, u);
    float mx  = wave_max(p);
    float sum = wave_sum(__expf(p - mx));
    float lsm = p - mx - __logf(sum);

    // stable descending top-16 selection (16 rounds of argmax + knockout)
    float val = p;
    int   selIdx = 0; float selLog = 0.f;
#pragma unroll
    for (int n = 0; n < NUM_SAMPLE; ++n) {
        float bv = val; int bi = lane;
        wave_argmax(bv, bi);
        float wl = __shfl(lsm, bi);
        if (lane == n) { selIdx = bi; selLog = wl; }
        if (lane == bi) val = -3.4e38f;
    }
    int   c = 0; float rank = -3.4e38f; float myLog = 0.f;
    if (lane < NUM_SAMPLE) {
        c = adj[pe * EDGE_T + selIdx];
        if (c < 0 || c > ITEM_HI) {
            unsigned h = hashu((unsigned)(b * 131071 + lane * 8191 + step * 524287 + 7));
            c = (int)(h % (unsigned)N_ITEMS);
        }
        const float4* uv = (const float4*)(dis_u + (size_t)u * 64);
        const float4* iv = (const float4*)(dis_i + (size_t)c * 64);
        float r = 0.f;
#pragma unroll
        for (int d = 0; d < 16; ++d) {
            float4 a = uv[d], bb = iv[d];
            r += a.x * bb.x + a.y * bb.y + a.z * bb.z + a.w * bb.w;
        }
        rank = r; myLog = selLog;
    }
    float bv = rank; int bi = lane;
    wave_argmax(bv, bi);
    if (lane == 0) {
        int   gneg = __shfl(c, bi);
        float glog = __shfl(myLog, bi) + one_log[b];
        pos_next[b]                        = gneg;
        out[step * BATCH + b]              = (float)gneg;   // neg_list (2 x B)
        out[2 * BATCH + step * BATCH + b]  = glog;          // prob_list (2 x B)
    }
}

// =================================== host ========================================
extern "C" void kernel_launch(void* const* d_in, const int* in_sizes, int n_in,
                              void* d_out, int out_size, void* d_ws, size_t ws_size,
                              hipStream_t stream) {
    const int*   u_id  = (const int*)d_in[0];
    const int*   pos_i = (const int*)d_in[1];
    const int*   adj   = (const int*)d_in[2];
    const int*   edge  = (const int*)d_in[3];
    const float* x     = (const float*)d_in[4];
    const float* W1r   = (const float*)d_in[5];
    const float* W1n   = (const float*)d_in[6];
    const float* b1    = (const float*)d_in[7];
    const float* W2r   = (const float*)d_in[8];
    const float* W2n   = (const float*)d_in[9];
    const float* b2    = (const float*)d_in[10];
    const float* dis_u = (const float*)d_in[11];
    const float* dis_i = (const float*)d_in[12];
    float* out = (float*)d_out;

    // workspace carve-up (with deliberate buffer reuse; ~78 MB total)
    char* w = (char*)d_ws;
    float* aggr1 = (float*)w;                          // [N_ENT,64]; reused as gcn
    float* gcn   = aggr1;
    w += (size_t)N_ENT * 64 * 4;
    float* cnt = (float*)w;  w += (size_t)N_ENT * 4;
    unsigned short* xbf = (unsigned short*)w;          // reused as h1bf
    unsigned short* hbf = xbf;
    w += (size_t)N_ENT * 64 * 2;
    unsigned short* abf  = (unsigned short*)w;         // reused as a2bf
    unsigned short* a2bf = abf;
    w += (size_t)N_ENT * 64 * 2;
    float* h1    = (float*)w; w += (size_t)N_ENT * 32 * 4;
    float* aggr2 = (float*)w; w += (size_t)N_ENT * 32 * 4;
    unsigned short* p1r = (unsigned short*)w; w += 2048 * 2;   // 32B-aligned packs
    unsigned short* p1n = (unsigned short*)w; w += 2048 * 2;
    unsigned short* p2r = (unsigned short*)w; w += 2048 * 2;
    unsigned short* p2n = (unsigned short*)w; w += 2048 * 2;
    int*   one_hop = (int*)w;   w += BATCH * 4;
    float* one_log = (float*)w; w += BATCH * 4;
    int*   pos_cur = (int*)w;   w += BATCH * 4;

    hipMemsetAsync(aggr1, 0, (size_t)N_ENT * 64 * 4, stream);
    hipMemsetAsync(cnt,   0, (size_t)N_ENT * 4,      stream);
    hipMemsetAsync(aggr2, 0, (size_t)N_ENT * 32 * 4, stream);

    prepB<<<8, 256, 0, stream>>>(W1r, W1n, W2r, W2n, p1r, p1n, p2r, p2n);

    // layer 1
    scatter64<<<(N_ENT * EDGE_T * 64) / 256, 256, 0, stream>>>(x, edge, aggr1, cnt);
    cvt64<<<(N_ENT * 64) / 256, 256, 0, stream>>>(x, aggr1, cnt, xbf, abf);
    int gemmBlocks = ((N_ENT / 16) * 32 + 255) / 256;
    gemm1<<<gemmBlocks, 256, 0, stream>>>(xbf, abf, p1r, p1n, b1, h1);
    // layer 2
    scatter32<<<(N_ENT * EDGE_T * 32) / 256, 256, 0, stream>>>(h1, edge, aggr2);
    cvt32<<<(N_ENT * 32) / 256, 256, 0, stream>>>(h1, aggr2, cnt, hbf, a2bf);
    gemm2<<<gemmBlocks, 256, 0, stream>>>(hbf, a2bf, p2r, p2n, b2, gcn);

    // sampling chain, K_STEP = 2
    int kgBlocks = (BATCH * 32) / 256;
    kg1<<<kgBlocks, 256, 0, stream>>>(gcn, adj, pos_i, u_id, one_hop, one_log);
    kg2<<<kgBlocks, 256, 0, stream>>>(gcn, adj, one_hop, u_id, dis_u, dis_i,
                                      one_log, 0, pos_cur, out);
    kg1<<<kgBlocks, 256, 0, stream>>>(gcn, adj, pos_cur, u_id, one_hop, one_log);
    kg2<<<kgBlocks, 256, 0, stream>>>(gcn, adj, one_hop, u_id, dis_u, dis_i,
                                      one_log, 1, pos_cur, out);

    (void)in_sizes; (void)n_in; (void)out_size; (void)ws_size;
}
